// MLSTM3_77601469104843
// MI455X (gfx1250) — matrix operand order repaired and checked by reading, not compiled, not run on hardware
//
#include <hip/hip_runtime.h>
#include <hip/hip_bf16.h>
#include <stdint.h>

// ---------------------------------------------------------------------------
// MLSTM (4-layer bilinear-gate LSTM), all dims 256, BATCH=64, G=4.
// bf16 WMMA (v_wmma_f32_16x16x32_bf16) two-stage GEMM pipeline.
// Stage 2 uses the Tensor Data Mover (tensor_load_to_lds) with a
// double-buffered LDS pipeline; fragments are read from LDS (ds_load_b128).
// ---------------------------------------------------------------------------

typedef __attribute__((ext_vector_type(16))) __bf16 v16bf;
typedef __attribute__((ext_vector_type(8)))  float  v8f;
typedef __attribute__((ext_vector_type(4)))  unsigned int v4u;
typedef __attribute__((ext_vector_type(8)))  int    v8i;
typedef __attribute__((ext_vector_type(4)))  int    v4i;

#define ELEMS 65536          // 256*256
#define LAYERS 4
#define GATES 4
#define BATCH 64

__device__ __forceinline__ unsigned short f32_to_bf16(float f) {
    unsigned int u = __float_as_uint(f);
    unsigned int r = u + 0x7FFFu + ((u >> 16) & 1u);   // round-to-nearest-even
    return (unsigned short)(r >> 16);
}

__device__ __forceinline__ v8f v8f_zero() {
    v8f z = {0.f, 0.f, 0.f, 0.f, 0.f, 0.f, 0.f, 0.f};
    return z;
}

union FragU {
    uint4 q[2];
    v16bf v;
};

// A-matrix fragment (16x32 bf16, M x K), row-major source, lda in elements.
// Layout: lane(0-15) row M=lane, K = kb+0..15;
//         lane(16-31) row M=lane-16, K = kb+16..31.
__device__ __forceinline__ v16bf load_a_frag(const unsigned short* __restrict__ A,
                                             int lda, int mbase, int kb) {
    int lane = threadIdx.x & 31;
    int row  = mbase + (lane & 15);
    int k0   = kb + ((lane >> 4) << 4);
    const unsigned short* p = A + (size_t)row * lda + k0;
    FragU f;
    f.q[0] = *(const uint4*)(p);        // K = k0 .. k0+7
    f.q[1] = *(const uint4*)(p + 8);    // K = k0+8 .. k0+15
    return f.v;
}

// B-matrix fragment (32x16 bf16, K x N) loaded from an N-major (pre-transposed)
// buffer Bt where Bt[n*ldb + k] = B[k][n].
// Layout (same K order as A): lane(0-15) col N=lane, K = kb+0..15; lane(16-31) K = kb+16..31.
__device__ __forceinline__ v16bf load_b_frag(const unsigned short* __restrict__ Bt,
                                             int ldb, int nbase, int kb) {
    int lane = threadIdx.x & 31;
    int col  = nbase + (lane & 15);
    int k0   = kb + ((lane >> 4) << 4);
    const unsigned short* p = Bt + (size_t)col * ldb + k0;
    FragU f;
    f.q[0] = *(const uint4*)(p);        // K = k0 .. k0+7
    f.q[1] = *(const uint4*)(p + 8);    // K = k0+8 .. k0+15
    return f.v;
}

// Same fragment layouts, but sourced from an LDS panel of shape [rows][32] bf16.
__device__ __forceinline__ v16bf lds_a_frag(const unsigned short* p0, int mloc) {
    int lane = threadIdx.x & 31;
    int row  = mloc + (lane & 15);
    int k0   = (lane >> 4) << 4;
    const unsigned short* p = p0 + row * 32 + k0;
    FragU f;
    f.q[0] = *(const uint4*)(p);
    f.q[1] = *(const uint4*)(p + 8);
    return f.v;
}

__device__ __forceinline__ v16bf lds_b_frag(const unsigned short* p0, int nloc) {
    int lane = threadIdx.x & 31;
    int col  = nloc + (lane & 15);
    int k0   = (lane >> 4) << 4;
    const unsigned short* p = p0 + col * 32 + k0;
    FragU f;
    f.q[0] = *(const uint4*)(p);
    f.q[1] = *(const uint4*)(p + 8);
    return f.v;
}

__device__ __forceinline__ v8f wmma_bf16(v16bf a, v16bf b, v8f c) {
    return __builtin_amdgcn_wmma_f32_16x16x32_bf16(
        /*neg_a=*/false, a, /*neg_b=*/false, b,
        /*c_mod=*/(short)0, c, /*reuse_a=*/false, /*reuse_b=*/false);
}

// ---------------------------------------------------------------------------
// Tensor Data Mover: 2D tile load (32 contiguous bf16 along K x 64 rows)
// from a row-major global matrix (row stride = tensor_row_len elements)
// into a compact LDS panel [64][32] bf16. Descriptor per ISA 8.3-8.6:
//   group0: count=1 | lds_addr | global_addr (57b) | type=2
//   group1: data_size=2B, tensor_dim0=row_len, tensor_dim1=256,
//           tile_dim0=32, tile_dim1=64, tensor_dim0_stride=row_len
//   groups 2/3: zero (2D tile)
// Toolchain here exposes the 6-arg builtin:
//   (uint32x4 g0, int32x8 g1, int32x4, int32x4, int32x8, i32 cpol)
// ---------------------------------------------------------------------------
__device__ __forceinline__ void tdm_load_tile(const unsigned short* gsrc,
                                              unsigned lds_off,
                                              int row_len) {
    unsigned long long ga = (unsigned long long)(uintptr_t)gsrc;
    v4u g0;
    g0.x = 1u;                                              // count=1
    g0.y = lds_off;                                         // lds_addr (bytes)
    g0.z = (unsigned)(ga & 0xffffffffu);                    // global_addr lo
    g0.w = (unsigned)((ga >> 32) & 0x01ffffffu) | 0x80000000u; // hi | type=2
    v8i g1;
    g1[0] = 0x00010000;                                     // data_size=1 (2B)
    g1[1] = (int)(((unsigned)row_len & 0xffffu) << 16);     // tensor_dim0 lo
    g1[2] = (int)((((unsigned)row_len >> 16) & 0xffffu) | (256u << 16)); // dim0 hi | dim1 lo
    g1[3] = (int)(32u << 16);                               // dim1 hi=0 | tile_dim0=32
    g1[4] = 64;                                             // tile_dim1=64 | tile_dim2=0
    g1[5] = row_len;                                        // tensor_dim0_stride lo
    g1[6] = 0;                                              // stride hi | dim1_stride lo
    g1[7] = 0;
    v4i gz4 = {0, 0, 0, 0};
    v8i gz8 = {0, 0, 0, 0, 0, 0, 0, 0};
    __builtin_amdgcn_tensor_load_to_lds(g0, g1, gz4, gz4, gz8, 0);
}

// ---------------------------------------------------------------------------
// Conversion kernels
// ---------------------------------------------------------------------------

__global__ __launch_bounds__(256) void convert_u_kernel(
    const float* __restrict__ Up, const float* __restrict__ Uq,
    const float* __restrict__ Ur,
    unsigned short* __restrict__ up, unsigned short* __restrict__ uq,
    unsigned short* __restrict__ ur, int n) {
    int i = blockIdx.x * 256 + threadIdx.x;
    if (i >= n) return;
    up[i] = f32_to_bf16(Up[i]);
    uq[i] = f32_to_bf16(Uq[i]);
    ur[i] = f32_to_bf16(Ur[i]);
}

// Build Vt_cat[l][g][n][k], k in [0,1024): [Vp^T | Vq_top^T | Vq_bot^T | Vr^T]
__global__ __launch_bounds__(256) void convert_v_kernel(
    const float* __restrict__ Vp, const float* __restrict__ Vq,
    const float* __restrict__ Vr, unsigned short* __restrict__ vt, int n) {
    int i = blockIdx.x * 256 + threadIdx.x;
    if (i >= n) return;
    int k    = i & 1023;
    int nrow = (i >> 10) & 255;
    int lg   = i >> 18;             // l*4+g, 0..15
    float v;
    if (k < 256)       v = Vp[((size_t)lg * 256 + k) * 256 + nrow];
    else if (k < 768)  v = Vq[((size_t)lg * 512 + (k - 256)) * 256 + nrow];
    else               v = Vr[((size_t)lg * 256 + (k - 768)) * 256 + nrow];
    vt[i] = f32_to_bf16(v);
}

// fp32 [mat][256][256] -> bf16 transposed [mat][256][256] via LDS 32x32 tiles.
__global__ __launch_bounds__(256) void transpose_bf16_kernel(
    const float* __restrict__ src, unsigned short* __restrict__ dst) {
    __shared__ float tile[32][33];
    int mat = blockIdx.z;
    const float* s = src + (size_t)mat * ELEMS;
    unsigned short* d = dst + (size_t)mat * ELEMS;
    int tx = threadIdx.x;                 // 0..31
    int ty = threadIdx.y;                 // 0..7
    int cb = blockIdx.x * 32;
    int rb = blockIdx.y * 32;
#pragma unroll
    for (int k = 0; k < 32; k += 8)
        tile[ty + k][tx] = s[(size_t)(rb + ty + k) * 256 + cb + tx];
    __syncthreads();
#pragma unroll
    for (int k = 0; k < 32; k += 8)
        d[(size_t)(cb + ty + k) * 256 + rb + tx] = f32_to_bf16(tile[tx][ty + k]);
}

__global__ __launch_bounds__(256) void zero_bf16_kernel(unsigned short* __restrict__ p, int n) {
    int i = blockIdx.x * 256 + threadIdx.x;
    if (i < n) p[i] = 0;
}

// ---------------------------------------------------------------------------
// Stage 1: T_cat[g][b][i][t*256+n] = U_t[l,g] @ M_t[b]   (256x256 GEMM, bf16)
//   t=0: Up @ X      t=1: Uq @ H_prev[l]   t=2: Uq @ H_below   t=3: Ur @ E
// grid: x=2 (128-col tiles), y=4 (64-row tiles), z=1024 ((t*4+g)*64+b)
// block: 256 threads = 8 waves as 2x4; wave tile 32x32 (2x2 wmma)
// Direct-from-L2 fragment loads (working set per (t,g): ~256 KB, L2-resident).
// ---------------------------------------------------------------------------
__global__ __launch_bounds__(256) void mlstm_stage1(
    const unsigned short* __restrict__ Up_bf,
    const unsigned short* __restrict__ Uq_bf,
    const unsigned short* __restrict__ Ur_bf,
    const unsigned short* __restrict__ Xt,
    const unsigned short* __restrict__ Ht,
    const unsigned short* __restrict__ Hbt,
    const unsigned short* __restrict__ Et,
    unsigned short* __restrict__ T,
    int layer) {
    int z = blockIdx.z;
    int b = z & 63;
    int g = (z >> 6) & 3;
    int t = z >> 8;

    const unsigned short* A;
    const unsigned short* Bt;
    size_t lgE = (size_t)(layer * GATES + g) * ELEMS;
    if (t == 0)      { A = Up_bf + lgE; Bt = Xt + (size_t)b * ELEMS; }
    else if (t == 1) { A = Uq_bf + lgE; Bt = Ht + (size_t)(layer * BATCH + b) * ELEMS; }
    else if (t == 2) { A = Uq_bf + lgE; Bt = Hbt + (size_t)b * ELEMS; }
    else             { A = Ur_bf + lgE; Bt = Et + (size_t)b * ELEMS; }

    int wave = threadIdx.x >> 5;
    int wr = wave >> 2;                        // 0..1
    int wc = wave & 3;                         // 0..3
    int mbase = blockIdx.y * 64 + wr * 32;
    int nbase = blockIdx.x * 128 + wc * 32;

    v8f acc00 = v8f_zero(), acc01 = v8f_zero();
    v8f acc10 = v8f_zero(), acc11 = v8f_zero();

    for (int kb = 0; kb < 256; kb += 32) {
        v16bf a0 = load_a_frag(A, 256, mbase, kb);
        v16bf a1 = load_a_frag(A, 256, mbase + 16, kb);
        v16bf b0 = load_b_frag(Bt, 256, nbase, kb);
        v16bf b1 = load_b_frag(Bt, 256, nbase + 16, kb);
        acc00 = wmma_bf16(a0, b0, acc00);
        acc01 = wmma_bf16(a0, b1, acc01);
        acc10 = wmma_bf16(a1, b0, acc10);
        acc11 = wmma_bf16(a1, b1, acc11);
    }

    // store bf16 into T row-major [g*64+b][256][1024], column block t*256
    unsigned short* Td = T + (size_t)(g * BATCH + b) * 256 * 1024 + (size_t)t * 256;
    int lane = threadIdx.x & 31;
    int cl = lane & 15;
    int rh = (lane >> 4) << 3;                 // +8 rows for hi lanes
    v8f accs[2][2] = {{acc00, acc01}, {acc10, acc11}};
#pragma unroll
    for (int mi = 0; mi < 2; ++mi)
#pragma unroll
        for (int ni = 0; ni < 2; ++ni)
#pragma unroll
            for (int r = 0; r < 8; ++r) {
                int row = mbase + mi * 16 + rh + r;
                int col = nbase + ni * 16 + cl;
                Td[(size_t)row * 1024 + col] = f32_to_bf16(accs[mi][ni][r]);
            }
}

// ---------------------------------------------------------------------------
// Stage 2: gates[g,b] = T_cat[g,b] (256x1024) @ V_cat[l,g] (1024x256) + Bias,
// then sigmoid + LSTM cell update, all 4 gates in registers per wave.
// grid: x=4 (64-col tiles), y=4 (64-row tiles), z=64 (batch)
// block: 256 threads = 8 waves as 2x4; wave tile 32x16 per gate (2 wmma)
// Double-buffered TDM pipeline: wave 0 issues tensor_load_to_lds for the
// 8 panels (A,B x 4 gates) of the next K-step, waits TENSORcnt, barrier.
// ---------------------------------------------------------------------------
__global__ __launch_bounds__(256) void mlstm_stage2(
    const unsigned short* __restrict__ T,    // [G*64][256][1024]
    const unsigned short* __restrict__ Vt,   // [L][G][256][1024]
    const float* __restrict__ Bias,          // [L][G][256][256]
    const float* __restrict__ Cprev,         // [L][64][256][256]
    float* __restrict__ Hout,                // layer base, [64][256][256]
    float* __restrict__ Cout,                // layer base
    unsigned short* __restrict__ Hbt,        // next layer H_below, transposed bf16
    int layer) {
    __shared__ unsigned short bufA[2][GATES][64 * 32];   // 32 KB
    __shared__ unsigned short bufB[2][GATES][64 * 32];   // 32 KB

    int b = blockIdx.z;
    int wave = threadIdx.x >> 5;
    int wr = wave >> 2;                       // 0..1  (32-row blocks)
    int wc = wave & 3;                        // 0..3  (16-col blocks)
    int mloc = wr * 32;
    int nloc = wc * 16;
    int mbase = blockIdx.y * 64 + mloc;
    int nbase = blockIdx.x * 64 + nloc;
    bool issuer = (wave == 0);

    // Global tile-row starts (element offsets advance by 32 along K each step).
    const unsigned short* Ag[GATES];
    const unsigned short* Bg[GATES];
#pragma unroll
    for (int g = 0; g < GATES; ++g) {
        Ag[g] = T + (size_t)(g * BATCH + b) * 256 * 1024 + (size_t)blockIdx.y * 64 * 1024;
        Bg[g] = Vt + (size_t)(layer * GATES + g) * 256 * 1024 + (size_t)blockIdx.x * 64 * 1024;
    }

    // Prologue: fill buffer 0 with K-step 0.
    if (issuer) {
#pragma unroll
        for (int g = 0; g < GATES; ++g) {
            tdm_load_tile(Ag[g], (unsigned)(uintptr_t)&bufA[0][g][0], 1024);
            tdm_load_tile(Bg[g], (unsigned)(uintptr_t)&bufB[0][g][0], 1024);
        }
        __builtin_amdgcn_s_wait_tensorcnt(0);
    }
    __syncthreads();

    v8f acc[GATES][2];
#pragma unroll
    for (int g = 0; g < GATES; ++g) {
        acc[g][0] = v8f_zero();
        acc[g][1] = v8f_zero();
    }

    for (int it = 0; it < 32; ++it) {         // K-steps of 32, K total = 1024
        int cur = it & 1;
        int nxt = cur ^ 1;
        bool more = (it + 1) < 32;
        if (issuer && more) {
            int koff = (it + 1) * 32;
#pragma unroll
            for (int g = 0; g < GATES; ++g) {
                tdm_load_tile(Ag[g] + koff, (unsigned)(uintptr_t)&bufA[nxt][g][0], 1024);
                tdm_load_tile(Bg[g] + koff, (unsigned)(uintptr_t)&bufB[nxt][g][0], 1024);
            }
        }
#pragma unroll
        for (int g = 0; g < GATES; ++g) {
            v16bf a0 = lds_a_frag(&bufA[cur][g][0], mloc);
            v16bf a1 = lds_a_frag(&bufA[cur][g][0], mloc + 16);
            v16bf b0 = lds_b_frag(&bufB[cur][g][0], nloc);
            acc[g][0] = wmma_bf16(a0, b0, acc[g][0]);
            acc[g][1] = wmma_bf16(a1, b0, acc[g][1]);
        }
        if (issuer && more) __builtin_amdgcn_s_wait_tensorcnt(0);
        __syncthreads();
    }

    // Elementwise combine: C/D layout is identical for all 4 gates, so each
    // lane owns the same (row,col) across gates -> pure register math.
    int lane = threadIdx.x & 31;
    int cl = lane & 15;
    int rh = (lane >> 4) << 3;
    size_t bias_base = (size_t)layer * GATES * ELEMS;
    size_t cp_base   = (size_t)(layer * BATCH + b) * ELEMS;
    size_t hb_base   = (size_t)b * ELEMS;

#pragma unroll
    for (int mi = 0; mi < 2; ++mi)
#pragma unroll
        for (int r = 0; r < 8; ++r) {
            int row = mbase + mi * 16 + rh + r;
            int col = nbase + cl;
            size_t off = (size_t)row * 256 + col;
            float xi = acc[0][mi][r] + Bias[bias_base + 0 * ELEMS + off];
            float xf = acc[1][mi][r] + Bias[bias_base + 1 * ELEMS + off];
            float xo = acc[2][mi][r] + Bias[bias_base + 2 * ELEMS + off];
            float xc = acc[3][mi][r] + Bias[bias_base + 3 * ELEMS + off];
            float I  = 1.0f / (1.0f + __expf(-xi));
            float F  = 1.0f / (1.0f + __expf(-xf));
            float O  = 1.0f / (1.0f + __expf(-xo));
            float Ch = 1.0f / (1.0f + __expf(-xc));
            float Cv = F * Cprev[cp_base + off] + I * Ch;
            float Hv = O * tanhf(Cv);
            Hout[hb_base + off] = Hv;
            Cout[hb_base + off] = Cv;
            // transposed bf16 H_below for the next layer's stage 1
            Hbt[hb_base + (size_t)col * 256 + row] = f32_to_bf16(Hv);
        }
}

// ---------------------------------------------------------------------------
// Launcher
// ---------------------------------------------------------------------------
extern "C" void kernel_launch(void* const* d_in, const int* in_sizes, int n_in,
                              void* d_out, int out_size, void* d_ws, size_t ws_size,
                              hipStream_t stream) {
    (void)in_sizes; (void)n_in; (void)out_size; (void)ws_size;

    const float* X  = (const float*)d_in[0];   // [64,256,256]
    const float* Hp = (const float*)d_in[1];   // [4,64,256,256]
    const float* Ex = (const float*)d_in[2];   // [64,256,256]
    const float* Cp = (const float*)d_in[3];   // [4,64,256,256]
    const float* Up = (const float*)d_in[4];   // [4,4,256,256]
    const float* Vp = (const float*)d_in[5];   // [4,4,256,256]
    const float* Uq = (const float*)d_in[6];   // [4,4,256,256]
    const float* Vq = (const float*)d_in[7];   // [4,4,512,256]
    const float* Ur = (const float*)d_in[8];   // [4,4,256,256]
    const float* Vr = (const float*)d_in[9];   // [4,4,256,256]
    const float* Bi = (const float*)d_in[10];  // [4,4,256,256]
    float* out = (float*)d_out;                // [2,4,64,256,256]

    char* ws = (char*)d_ws;
    size_t off = 0;
    auto walloc = [&](size_t bytes) -> char* {
        char* p = ws + off;
        off += (bytes + 255) & ~(size_t)255;
        return p;
    };
    unsigned short* Xt   = (unsigned short*)walloc((size_t)BATCH * ELEMS * 2);
    unsigned short* Et   = (unsigned short*)walloc((size_t)BATCH * ELEMS * 2);
    unsigned short* Ht   = (unsigned short*)walloc((size_t)LAYERS * BATCH * ELEMS * 2);
    unsigned short* Hbt  = (unsigned short*)walloc((size_t)BATCH * ELEMS * 2);
    unsigned short* UpB  = (unsigned short*)walloc((size_t)LAYERS * GATES * ELEMS * 2);
    unsigned short* UqB  = (unsigned short*)walloc((size_t)LAYERS * GATES * ELEMS * 2);
    unsigned short* UrB  = (unsigned short*)walloc((size_t)LAYERS * GATES * ELEMS * 2);
    unsigned short* VtB  = (unsigned short*)walloc((size_t)LAYERS * GATES * 256 * 1024 * 2);
    unsigned short* Tbuf = (unsigned short*)walloc((size_t)GATES * BATCH * 256 * 1024 * 2);

    // --- conversions (every call; deterministic) ---
    {
        int n = LAYERS * GATES * ELEMS;                 // 1,048,576
        convert_u_kernel<<<n / 256, 256, 0, stream>>>(Up, Uq, Ur, UpB, UqB, UrB, n);
        int nv = LAYERS * GATES * 256 * 1024;           // 4,194,304
        convert_v_kernel<<<nv / 256, 256, 0, stream>>>(Vp, Vq, Vr, VtB, nv);
    }
    {
        dim3 blk(32, 8);
        transpose_bf16_kernel<<<dim3(8, 8, BATCH), blk, 0, stream>>>(X, Xt);
        transpose_bf16_kernel<<<dim3(8, 8, BATCH), blk, 0, stream>>>(Ex, Et);
        transpose_bf16_kernel<<<dim3(8, 8, LAYERS * BATCH), blk, 0, stream>>>(Hp, Ht);
        int n = BATCH * ELEMS;
        zero_bf16_kernel<<<n / 256, 256, 0, stream>>>(Hbt, n);
    }

    // --- sequential layers ---
    for (int l = 0; l < LAYERS; ++l) {
        mlstm_stage1<<<dim3(2, 4, 1024), 256, 0, stream>>>(
            UpB, UqB, UrB, Xt, Ht, Hbt, Et, Tbuf, l);

        float* Hout = out + (size_t)(0 * LAYERS + l) * BATCH * ELEMS;
        float* Cout = out + (size_t)(1 * LAYERS + l) * BATCH * ELEMS;
        mlstm_stage2<<<dim3(4, 4, BATCH), 256, 0, stream>>>(
            Tbuf, VtB, Bi, Cp, Hout, Cout, Hbt, l);
    }
}
